// GroupDCNv3Layer_53541062312409
// MI455X (gfx1250) — matrix operand order repaired and checked by reading, not compiled, not run on hardware
//
#include <hip/hip_runtime.h>
#include <hip/hip_fp16.h>
#include <math.h>

typedef _Float16 v16h __attribute__((ext_vector_type(16)));
typedef _Float16 v8h  __attribute__((ext_vector_type(8)));
typedef float    v8f  __attribute__((ext_vector_type(8)));

#define NB     4
#define C_TOT  256
#define G_GRP  8
#define CG_    32
#define H_DIM  64
#define W_DIM  64
#define HW_    (H_DIM * W_DIM)          // 4096
#define NPOS   (NB * HW_)               // 16384
#define KDIM   256
#define NCOL   224                       // 144 offsets + 72 mask logits, padded to 224
#define NTILES_N 14                      // 224/16
#define NTILES_M (NPOS / 16)             // 1024

// ---------------------------------------------------------------------------
// 1/9. Generic tiled transpose: in is (R x Cc) per sample, out is (Cc x R).
// ---------------------------------------------------------------------------
__global__ void k_transpose(const float* __restrict__ in, float* __restrict__ out,
                            int R, int Cc) {
  __shared__ float tile[32][33];
  const float* inS  = in  + (size_t)blockIdx.z * R * Cc;
  float*       outS = out + (size_t)blockIdx.z * R * Cc;
  int c0 = blockIdx.x * 32;
  int r0 = blockIdx.y * 32;
  int tx = threadIdx.x, ty = threadIdx.y;
#pragma unroll
  for (int j = 0; j < 32; j += 8)
    tile[ty + j][tx] = inS[(size_t)(r0 + ty + j) * Cc + (c0 + tx)];
  __syncthreads();
#pragma unroll
  for (int j = 0; j < 32; j += 8)
    outS[(size_t)(c0 + ty + j) * R + (r0 + tx)] = tile[tx][ty + j];
}

// ---------------------------------------------------------------------------
// 2. Depthwise 3x3 conv in NHWC + bias; per-block sum/sumsq partials.
//    One block == one (n,h,w) position, 256 channels (coalesced).
// ---------------------------------------------------------------------------
__global__ void k_dwconv(const float* __restrict__ x_nhwc,
                         const float* __restrict__ dw_w,
                         const float* __restrict__ dw_b,
                         float* __restrict__ conv1,
                         float* __restrict__ partials) {
  int idx = blockIdx.x * 256 + threadIdx.x;
  int c =  idx        & 255;
  int w = (idx >> 8)  & 63;
  int h = (idx >> 14) & 63;
  int n =  idx >> 20;
  const float* xs = x_nhwc + (size_t)n * HW_ * C_TOT;
  float acc = dw_b[c];
#pragma unroll
  for (int ky = 0; ky < 3; ++ky) {
    int hy = h + ky - 1;
    if (hy < 0 || hy >= H_DIM) continue;
#pragma unroll
    for (int kx = 0; kx < 3; ++kx) {
      int wx = w + kx - 1;
      if (wx < 0 || wx >= W_DIM) continue;
      acc += xs[((size_t)hy * W_DIM + wx) * C_TOT + c] * dw_w[c * 9 + ky * 3 + kx];
    }
  }
  conv1[idx] = acc;

  __shared__ float s1[256], s2[256];
  s1[threadIdx.x] = acc;
  s2[threadIdx.x] = acc * acc;
  __syncthreads();
  for (int s = 128; s > 0; s >>= 1) {
    if (threadIdx.x < s) {
      s1[threadIdx.x] += s1[threadIdx.x + s];
      s2[threadIdx.x] += s2[threadIdx.x + s];
    }
    __syncthreads();
  }
  if (threadIdx.x == 0) {
    partials[blockIdx.x * 2 + 0] = s1[0];
    partials[blockIdx.x * 2 + 1] = s2[0];
  }
}

// ---------------------------------------------------------------------------
// 3. Deterministic stats reduction: one block per sample, fixed order.
// ---------------------------------------------------------------------------
__global__ void k_stats(const float* __restrict__ partials, float* __restrict__ stats) {
  int n = blockIdx.x;
  float s = 0.f, ss = 0.f;
  for (int i = threadIdx.x; i < HW_; i += 256) {   // 4096 blocks/sample
    s  += partials[(n * HW_ + i) * 2 + 0];
    ss += partials[(n * HW_ + i) * 2 + 1];
  }
  __shared__ float s1[256], s2[256];
  s1[threadIdx.x] = s;
  s2[threadIdx.x] = ss;
  __syncthreads();
  for (int st = 128; st > 0; st >>= 1) {
    if (threadIdx.x < st) {
      s1[threadIdx.x] += s1[threadIdx.x + st];
      s2[threadIdx.x] += s2[threadIdx.x + st];
    }
    __syncthreads();
  }
  if (threadIdx.x == 0) {
    const float invN = 1.f / (float)(C_TOT * HW_);
    float mean = s1[0] * invN;
    float var  = s2[0] * invN - mean * mean;
    stats[n * 2 + 0] = mean;
    stats[n * 2 + 1] = rsqrtf(var + 1e-5f);
  }
}

// ---------------------------------------------------------------------------
// 4. GroupNorm affine + exact GELU -> f16 NHWC activations for the WMMA GEMM.
// ---------------------------------------------------------------------------
__global__ void k_normgelu(const float* __restrict__ conv1,
                           const float* __restrict__ stats,
                           const float* __restrict__ gn_g,
                           const float* __restrict__ gn_b,
                           _Float16* __restrict__ x1h) {
  int idx = blockIdx.x * 256 + threadIdx.x;
  int c = idx & 255;
  int n = idx >> 20;
  float mean = stats[n * 2 + 0];
  float rstd = stats[n * 2 + 1];
  float y = (conv1[idx] - mean) * rstd * gn_g[c] + gn_b[c];
  float g = 0.5f * y * (1.f + erff(y * 0.70710678118654752f));
  x1h[idx] = (_Float16)g;
}

// ---------------------------------------------------------------------------
// 5. Pack [off_w | mask_w | pad] -> K-major f16 weight (224 cols x 256 K),
//    plus fused bias vector.
// ---------------------------------------------------------------------------
__global__ void k_prepw(const float* __restrict__ off_w, const float* __restrict__ off_b,
                        const float* __restrict__ mask_w, const float* __restrict__ mask_b,
                        _Float16* __restrict__ wT, float* __restrict__ bcomb) {
  int idx = blockIdx.x * 256 + threadIdx.x;
  if (idx < NCOL * KDIM) {
    int nn = idx / KDIM;
    int k  = idx % KDIM;
    float v = 0.f;
    if (nn < 144)      v = off_w[k * 144 + nn];
    else if (nn < 216) v = mask_w[k * 72 + (nn - 144)];
    wT[idx] = (_Float16)v;
  } else {
    int j = idx - NCOL * KDIM;
    if (j < NCOL) {
      float v = 0.f;
      if (j < 144)      v = off_b[j];
      else if (j < 216) v = mask_b[j - 144];
      bcomb[j] = v;
    }
  }
}

// ---------------------------------------------------------------------------
// 6. WMMA GEMM: (16384 x 256) f16 @ (256 x 224) f16 -> f32 + bias.
//    One wave per 16x16 tile; K=256 -> 8x v_wmma_f32_16x16x32_f16.
//    A layout (16-bit A 16x32): lane&15 = row M, lane>>4 selects
//    K 0..15 vs 16..31 (contiguous, same as B). B: lane&15 = col N, lane>>4
//    selects K 0..15 vs 16..31 -> contiguous 16-half load from K-major weights.
// ---------------------------------------------------------------------------
__global__ void k_gemm(const _Float16* __restrict__ A, const _Float16* __restrict__ BT,
                       const float* __restrict__ bias, float* __restrict__ Dout) {
  int wave = (blockIdx.x * blockDim.x + threadIdx.x) >> 5;
  int lane = threadIdx.x & 31;
  int tileN = wave % NTILES_N;
  int tileM = wave / NTILES_N;
  int half = lane >> 4;
  int l15  = lane & 15;
  int m    = tileM * 16 + l15;
  int ncol = tileN * 16 + l15;
  const _Float16* Arow = A  + (size_t)m * KDIM;
  const _Float16* Brow = BT + (size_t)ncol * KDIM;
  const int kb8  = half * 8;
  const int kb16 = half * 16;
  v8f acc = {};
#pragma unroll
  for (int k0 = 0; k0 < KDIM; k0 += 32) {
    if (k0 + 64 < KDIM) __builtin_prefetch(Arow + k0 + 64, 0, 1);
    v16h a = *(const v16h*)(Arow + k0 + kb16);
    v16h b = *(const v16h*)(Brow + k0 + kb16);
    acc = __builtin_amdgcn_wmma_f32_16x16x32_f16(false, a, false, b,
                                                 (short)0, acc, false, false);
  }
  // D layout: VGPR r -> row (half*8 + r), col = lane&15.
  int rowBase = tileM * 16 + half * 8;
  float bv = bias[ncol];
#pragma unroll
  for (int r = 0; r < 8; ++r)
    Dout[(size_t)(rowBase + r) * NCOL + ncol] = acc[r] + bv;
}

// ---------------------------------------------------------------------------
// 7. Softmax over the 9 sampling points per (position, group), in place.
// ---------------------------------------------------------------------------
__global__ void k_softmax(float* __restrict__ offmask) {
  int idx = blockIdx.x * 256 + threadIdx.x;   // NPOS * G
  int g   = idx & 7;
  int row = idx >> 3;
  float* m = offmask + (size_t)row * NCOL + 144 + g * 9;
  float v[9];
  float mx = -3.402823e38f;
#pragma unroll
  for (int p = 0; p < 9; ++p) { v[p] = m[p]; mx = fmaxf(mx, v[p]); }
  float s = 0.f;
#pragma unroll
  for (int p = 0; p < 9; ++p) { v[p] = expf(v[p] - mx); s += v[p]; }
  float inv = 1.f / s;
#pragma unroll
  for (int p = 0; p < 9; ++p) m[p] = v[p] * inv;
}

// ---------------------------------------------------------------------------
// 8. DCNv3 core. One wave per (position, group); lane = channel (CG=32 fits
//    wave32 exactly). Padded-coordinate algebra: px = 1 + w + kx + off_x,
//    nonzero only for padded coords in [1,64] (zero ring + zero OOB).
// ---------------------------------------------------------------------------
__global__ void k_dcnv3(const float* __restrict__ x_nhwc,
                        const float* __restrict__ offmask,
                        float* __restrict__ out_nhwc) {
  int lane  = threadIdx.x & 31;
  int gwave = (blockIdx.x * blockDim.x + threadIdx.x) >> 5;
  int g   = gwave & 7;
  int pos = gwave >> 3;
  int n  = pos >> 12;
  int hw = pos & 4095;
  int h = hw >> 6, w = hw & 63;
  const float* xs = x_nhwc + (size_t)n * HW_ * C_TOT + g * CG_ + lane;
  const float* om = offmask + (size_t)pos * NCOL;
  float acc = 0.f;
#pragma unroll
  for (int p = 0; p < 9; ++p) {
    float ox = om[(g * 9 + p) * 2 + 0];
    float oy = om[(g * 9 + p) * 2 + 1];
    float mw = om[144 + g * 9 + p];
    // p = iw*3 + ih; kvals = {-1,0,1}; Wp/Hp cancel exactly.
    float px = (float)(w + (p / 3)) + ox;   // = 1 + w + (p/3 - 1) + ox
    float py = (float)(h + (p % 3)) + oy;
    float x0f = floorf(px), y0f = floorf(py);
    int   x0  = (int)x0f,  y0  = (int)y0f;
    float wx = px - x0f, wy = py - y0f;
    float c00 = (1.f - wy) * (1.f - wx);
    float c01 = (1.f - wy) * wx;
    float c10 = wy * (1.f - wx);
    float c11 = wy * wx;
    float v00 = 0.f, v01 = 0.f, v10 = 0.f, v11 = 0.f;
    if ((unsigned)(y0 - 1) < 64u && (unsigned)(x0 - 1) < 64u)
      v00 = xs[((size_t)(y0 - 1) * W_DIM + (x0 - 1)) * C_TOT];
    if ((unsigned)(y0 - 1) < 64u && (unsigned)(x0) < 64u)
      v01 = xs[((size_t)(y0 - 1) * W_DIM + (x0)) * C_TOT];
    if ((unsigned)(y0) < 64u && (unsigned)(x0 - 1) < 64u)
      v10 = xs[((size_t)(y0) * W_DIM + (x0 - 1)) * C_TOT];
    if ((unsigned)(y0) < 64u && (unsigned)(x0) < 64u)
      v11 = xs[((size_t)(y0) * W_DIM + (x0)) * C_TOT];
    acc += mw * (c00 * v00 + c01 * v01 + c10 * v10 + c11 * v11);
  }
  out_nhwc[(size_t)pos * C_TOT + g * CG_ + lane] = acc;
}

// ---------------------------------------------------------------------------
// Launch
// ---------------------------------------------------------------------------
extern "C" void kernel_launch(void* const* d_in, const int* in_sizes, int n_in,
                              void* d_out, int out_size, void* d_ws, size_t ws_size,
                              hipStream_t stream) {
  (void)in_sizes; (void)n_in; (void)out_size; (void)ws_size;
  const float* x      = (const float*)d_in[0];
  const float* dw_w   = (const float*)d_in[1];
  const float* dw_b   = (const float*)d_in[2];
  const float* gn_g   = (const float*)d_in[3];
  const float* gn_b   = (const float*)d_in[4];
  const float* off_w  = (const float*)d_in[5];
  const float* off_b  = (const float*)d_in[6];
  const float* mask_w = (const float*)d_in[7];
  const float* mask_b = (const float*)d_in[8];
  float* out = (float*)d_out;

  char* ws = (char*)d_ws;
  float*    x_nhwc   = (float*)(ws + 0);                  // 16 MB
  float*    conv1    = (float*)(ws + 16777216);           // 16 MB (reused as out_nhwc)
  _Float16* x1h      = (_Float16*)(ws + 33554432);        //  8 MB
  float*    offmask  = (float*)(ws + 41943040);           // 14 MB (16384 x 224)
  _Float16* wcombT   = (_Float16*)(ws + 56623104);        // 112 KB
  float*    bcomb    = (float*)(ws + 56737792);           //  1 KB
  float*    partials = (float*)(ws + 56738816);           // 128 KB
  float*    stats    = (float*)(ws + 56869888);           // 32 B
  float*    out_nhwc = conv1;                              // conv1 dead after normgelu

  // 1. NCHW -> NHWC  (per sample: 256 x 4096 -> 4096 x 256)
  k_transpose<<<dim3(HW_ / 32, C_TOT / 32, NB), dim3(32, 8), 0, stream>>>(
      x, x_nhwc, C_TOT, HW_);
  // 2. depthwise conv + partial stats
  k_dwconv<<<NPOS, 256, 0, stream>>>(x_nhwc, dw_w, dw_b, conv1, partials);
  // 3. mean / rstd per sample (deterministic)
  k_stats<<<NB, 256, 0, stream>>>(partials, stats);
  // 4. norm + GELU -> f16
  k_normgelu<<<NPOS, 256, 0, stream>>>(conv1, stats, gn_g, gn_b, x1h);
  // 5. pack weights K-major + bias
  k_prepw<<<(NCOL * KDIM + 224 + 255) / 256, 256, 0, stream>>>(
      off_w, off_b, mask_w, mask_b, wcombT, bcomb);
  // 6. WMMA projection GEMM (offsets + mask logits fused)
  k_gemm<<<(NTILES_M * NTILES_N) / 4, 128, 0, stream>>>(x1h, wcombT, bcomb, offmask);
  // 7. softmax over 9 points per (pos, group)
  k_softmax<<<(NPOS * G_GRP) / 256, 256, 0, stream>>>(offmask);
  // 8. deformable sampling
  k_dcnv3<<<(NPOS * G_GRP) / 4, 128, 0, stream>>>(x_nhwc, offmask, out_nhwc);
  // 9. NHWC -> NCHW (per sample: 4096 x 256 -> 256 x 4096)
  k_transpose<<<dim3(C_TOT / 32, HW_ / 32, NB), dim3(32, 8), 0, stream>>>(
      out_nhwc, out, HW_, C_TOT);
}
